// Net_26414048870995
// MI455X (gfx1250) — matrix operand order repaired and checked by reading, not compiled, not run on hardware
//
#include <hip/hip_runtime.h>
#include <hip/hip_bf16.h>
#include <math.h>

// ---------------- problem constants ----------------
#define NATOM   300000
#define NEDGE   1200000
#define NBOND   300000
#define NANG    500000
#define NTOR    700000
#define NOF     600000
#define NNBP    2400000
#define NTYPES  10
#define DD      128
#define RU      32

typedef __attribute__((ext_vector_type(16))) _Float16 v16h;
typedef __attribute__((ext_vector_type(8)))  float    v8f;

__device__ __forceinline__ v8f wmma_f16(v16h a, v16h b, v8f c) {
  // (neg_a, A, neg_b, B, c_mod, C, reuse_a, reuse_b)
  return __builtin_amdgcn_wmma_f32_16x16x32_f16(false, a, false, b, (short)0, c, false, false);
}

// assign one h-row's 16 floats into this lane's A-fragment slots
__device__ __forceinline__ void loadrow16(float s[16], const float* __restrict__ p, int k0) {
  const float4* q0 = (const float4*)(p + k0);
  const float4* q1 = (const float4*)(p + k0 + 8);
  float4 a = q0[0], b = q0[1], c = q1[0], d = q1[1];
  s[0]=a.x;  s[1]=a.y;  s[2]=a.z;  s[3]=a.w;
  s[4]=b.x;  s[5]=b.y;  s[6]=b.z;  s[7]=b.w;
  s[8]=c.x;  s[9]=c.y;  s[10]=c.z; s[11]=c.w;
  s[12]=d.x; s[13]=d.y; s[14]=d.z; s[15]=d.w;
}

// accumulate one h-row's 16 floats into this lane's A-fragment slots
__device__ __forceinline__ void addrow16(float s[16], const float* __restrict__ p, int k0) {
  const float4* q0 = (const float4*)(p + k0);
  const float4* q1 = (const float4*)(p + k0 + 8);
  float4 a = q0[0], b = q0[1], c = q1[0], d = q1[1];
  s[0]+=a.x;  s[1]+=a.y;  s[2]+=a.z;  s[3]+=a.w;
  s[4]+=b.x;  s[5]+=b.y;  s[6]+=b.z;  s[7]+=b.w;
  s[8]+=c.x;  s[9]+=c.y;  s[10]+=c.z; s[11]+=c.w;
  s[12]+=d.x; s[13]+=d.y; s[14]+=d.z; s[15]+=d.w;
}

__device__ __forceinline__ v16h cvt16(const float s[16]) {
  v16h a;
  a[0]=(_Float16)s[0];   a[1]=(_Float16)s[1];   a[2]=(_Float16)s[2];   a[3]=(_Float16)s[3];
  a[4]=(_Float16)s[4];   a[5]=(_Float16)s[5];   a[6]=(_Float16)s[6];   a[7]=(_Float16)s[7];
  a[8]=(_Float16)s[8];   a[9]=(_Float16)s[9];   a[10]=(_Float16)s[10]; a[11]=(_Float16)s[11];
  a[12]=(_Float16)s[12]; a[13]=(_Float16)s[13]; a[14]=(_Float16)s[14]; a[15]=(_Float16)s[15];
  return a;
}

// ---------------- utility kernels ----------------
__global__ void zerof(float* __restrict__ p, long n) {
  long i = (long)blockIdx.x * 256 + threadIdx.x;
  if (i < n) p[i] = 0.0f;
}

// f32 KxN row-major -> f16 NxK row-major (column-major original) so B frags are contiguous
__global__ void transcvt(const float* __restrict__ src, _Float16* __restrict__ dst, int Ncol) {
  int t = blockIdx.x * 256 + threadIdx.x;
  if (t >= 128 * Ncol) return;
  int k = t / Ncol, n = t % Ncol;
  dst[(size_t)n * 128 + k] = (_Float16)src[(size_t)k * Ncol + n];
}

// one-hot segment_sum: cnt[dst][type[src]] += 1
__global__ void scatter_count(const int* __restrict__ type, const int* __restrict__ src,
                              const int* __restrict__ dst, float* __restrict__ cnt) {
  long e = (long)blockIdx.x * 256 + threadIdx.x;
  if (e >= NEDGE) return;
  atomicAdd(&cnt[(size_t)dst[e] * NTYPES + type[src[e]]], 1.0f);
}

// layer0: h = relu(cnt @ gW0 + gb0)   (cnt: Nx10, gW0: 10x128)
__global__ void layer0(const float* __restrict__ cnt, const float* __restrict__ W0,
                       const float* __restrict__ b0, float* __restrict__ h) {
  int i = blockIdx.x, d = threadIdx.x;   // grid=NATOM, block=128
  float s = b0[d];
  const float* c = cnt + (size_t)i * NTYPES;
#pragma unroll
  for (int t = 0; t < NTYPES; ++t) s += c[t] * W0[t * 128 + d];
  h[(size_t)i * 128 + d] = fmaxf(s, 0.0f);
}

// m[dst] += h[src]  (32 lanes per edge, 4 floats per lane)
__global__ void scatter_add(const float* __restrict__ hsrc, const int* __restrict__ src,
                            const int* __restrict__ dst, float* __restrict__ m) {
  long tid = (long)blockIdx.x * 256 + threadIdx.x;
  long e = tid >> 5;
  if (e >= NEDGE) return;
  int l = (int)(tid & 31);
  float4 v = *(const float4*)(hsrc + (size_t)src[e] * 128 + l * 4);
  float* d = m + (size_t)dst[e] * 128 + l * 4;
  atomicAdd(d + 0, v.x); atomicAdd(d + 1, v.y);
  atomicAdd(d + 2, v.z); atomicAdd(d + 3, v.w);
}

// ---------------- WMMA GEMM: H = (relu?)(A[N x 128] @ W[128x128] + bias) ----------------
// Block 256 = 8 waves; block covers 32 rows x 128 cols; wave = 16 rows x 32 cols
// NATOM % 32 == 0 so no row guards needed.
__global__ __launch_bounds__(256) void gemm_gcn(const float* __restrict__ A,
                                                const _Float16* __restrict__ Wt,
                                                const float* __restrict__ bias,
                                                float* __restrict__ H, int doRelu) {
  int tid = threadIdx.x;
  int wave = tid >> 5, lane = tid & 31;
  int lm = lane & 15, lh = lane >> 4;
  int rowBase = blockIdx.x * 32 + ((wave >> 2) << 4);
  int colBase = (wave & 3) << 5;
  const float* Arow = A + (size_t)(rowBase + lm) * 128;
  v8f acc0 = {}, acc1 = {};
#pragma unroll
  for (int kk = 0; kk < 128; kk += 32) {
    int k0 = kk + (lh << 4);
    float s[16];
    loadrow16(s, Arow, k0);
    v16h a = cvt16(s);
    v16h b0 = *(const v16h*)(Wt + (size_t)(colBase + lm) * 128 + kk + (lh << 4));
    v16h b1 = *(const v16h*)(Wt + (size_t)(colBase + 16 + lm) * 128 + kk + (lh << 4));
    acc0 = wmma_f16(a, b0, acc0);
    acc1 = wmma_f16(a, b1, acc1);
  }
#pragma unroll
  for (int r = 0; r < 8; ++r) {
    int row = rowBase + r + (lh << 3);
    int c0 = colBase + lm;
    float v0 = acc0[r] + bias[c0];
    float v1 = acc1[r] + bias[c0 + 16];
    if (doRelu) { v0 = fmaxf(v0, 0.0f); v1 = fmaxf(v1, 0.0f); }
    H[(size_t)row * 128 + c0] = v0;
    H[(size_t)row * 128 + c0 + 16] = v1;
  }
}

// ---------------- WMMA readout: t=tanh(hsum@W1+b1); out=|t@W2+b2| ----------------
// mode 1: hsum = h[row]
// mode 2: hsum = h[i0[2r]] + h[i0[2r+1]]
// mode 3: hsum = h[i0[r]] + h[i1[2r]] + h[i1[2r+1]]
// mode 4: hsum = sum_{j<4} h[i0[4r+j]]
// Block 256 = 8 waves x 16 rows = 128 rows/block.
__global__ __launch_bounds__(256) void readout(const float* __restrict__ h,
                                               const int* __restrict__ i0,
                                               const int* __restrict__ i1,
                                               int mode, int total,
                                               const _Float16* __restrict__ W1t,
                                               const float* __restrict__ b1,
                                               const float* __restrict__ W2,
                                               const float* __restrict__ b2,
                                               float* __restrict__ outK,
                                               float* __restrict__ outEq) {
  __shared__ float lds[8][16][32];
  int tid = threadIdx.x;
  int wave = tid >> 5, lane = tid & 31;
  int lm = lane & 15, lh = lane >> 4;
  int gr = blockIdx.x * 128 + wave * 16 + lm;
  if (gr >= total) gr = 0;   // clamp: compute garbage, don't store (EXEC stays full)

  const float* p0; const float* p1 = nullptr; const float* p2 = nullptr; const float* p3 = nullptr;
  if (mode == 1)      { p0 = h + (size_t)gr * 128; }
  else if (mode == 2) { p0 = h + (size_t)i0[2 * gr] * 128;
                        p1 = h + (size_t)i0[2 * gr + 1] * 128; }
  else if (mode == 3) { p0 = h + (size_t)i0[gr] * 128;
                        p1 = h + (size_t)i1[2 * gr] * 128;
                        p2 = h + (size_t)i1[2 * gr + 1] * 128; }
  else                { p0 = h + (size_t)i0[4 * gr] * 128;
                        p1 = h + (size_t)i0[4 * gr + 1] * 128;
                        p2 = h + (size_t)i0[4 * gr + 2] * 128;
                        p3 = h + (size_t)i0[4 * gr + 3] * 128; }

  v8f acc0 = {}, acc1 = {};
#pragma unroll
  for (int kk = 0; kk < 128; kk += 32) {
    int k0 = kk + (lh << 4);
    float s[16];
    loadrow16(s, p0, k0);
    if (mode >= 2) addrow16(s, p1, k0);
    if (mode >= 3) addrow16(s, p2, k0);
    if (mode >= 4) addrow16(s, p3, k0);
    v16h a = cvt16(s);
    v16h b0 = *(const v16h*)(W1t + (size_t)lm * 128 + kk + (lh << 4));
    v16h b1 = *(const v16h*)(W1t + (size_t)(16 + lm) * 128 + kk + (lh << 4));
    acc0 = wmma_f16(a, b0, acc0);
    acc1 = wmma_f16(a, b1, acc1);
  }
#pragma unroll
  for (int r = 0; r < 8; ++r) {
    int mrow = r + (lh << 3);
    lds[wave][mrow][lm]      = tanhf(acc0[r] + b1[lm]);
    lds[wave][mrow][lm + 16] = tanhf(acc1[r] + b1[lm + 16]);
  }
  __syncthreads();
  // 256 threads -> (8 waves x 16 rows) x 2 outputs
  int slot = tid >> 5;
  int rl = (tid & 31) >> 1;
  int o = tid & 1;
  int row = blockIdx.x * 128 + slot * 16 + rl;
  if (row < total) {
    float s = b2[o];
#pragma unroll
    for (int j = 0; j < RU; ++j) s += lds[slot][rl][j] * W2[2 * j + o];
    (o == 0 ? outK : outEq)[row] = fabsf(s);
  }
}

// ---------------- pair mixing ----------------
__global__ void pair_lj(const float* __restrict__ kAtom, const float* __restrict__ eqAtom,
                        const int* __restrict__ pairs, long M,
                        float* __restrict__ sig, float* __restrict__ eps) {
  long i = (long)blockIdx.x * 256 + threadIdx.x;
  if (i >= M) return;
  int a = pairs[2 * i], b = pairs[2 * i + 1];
  sig[i] = 0.5f * (kAtom[a] + kAtom[b]);
  eps[i] = sqrtf(eqAtom[a] * eqAtom[b]);
}

// ---------------- host-side orchestration ----------------
extern "C" void kernel_launch(void* const* d_in, const int* in_sizes, int n_in,
                              void* d_out, int out_size, void* d_ws, size_t ws_size,
                              hipStream_t stream) {
  const int* atom_type  = (const int*)d_in[0];
  const int* edge_src   = (const int*)d_in[1];
  const int* edge_dst   = (const int*)d_in[2];
  const int* bond_atoms = (const int*)d_in[3];
  const int* ang_center = (const int*)d_in[4];
  const int* ang_sides  = (const int*)d_in[5];
  const int* tor_atoms  = (const int*)d_in[6];
  const int* of_atoms   = (const int*)d_in[7];
  const int* nb_atoms   = (const int*)d_in[8];
  const float* gW0 = (const float*)d_in[9];
  const float* gb0 = (const float*)d_in[10];
  const float* gW1 = (const float*)d_in[11];
  const float* gb1 = (const float*)d_in[12];
  const float* gW2 = (const float*)d_in[13];
  const float* gb2 = (const float*)d_in[14];
  const float* aW1 = (const float*)d_in[15];
  const float* ab1 = (const float*)d_in[16];
  const float* aW2 = (const float*)d_in[17];
  const float* ab2 = (const float*)d_in[18];
  const float* bW1 = (const float*)d_in[19];
  const float* bb1 = (const float*)d_in[20];
  const float* bW2 = (const float*)d_in[21];
  const float* bb2 = (const float*)d_in[22];
  const float* nW1 = (const float*)d_in[23];
  const float* nb1 = (const float*)d_in[24];
  const float* nW2 = (const float*)d_in[25];
  const float* nb2 = (const float*)d_in[26];
  const float* tW1 = (const float*)d_in[27];
  const float* tb1 = (const float*)d_in[28];
  const float* tW2 = (const float*)d_in[29];
  const float* tb2 = (const float*)d_in[30];

  float* out = (float*)d_out;

  // workspace layout
  float* m = (float*)d_ws;                       // NATOM*128 f32 (also aliases cnt NATOM*10)
  float* h = m + (size_t)NATOM * 128;            // NATOM*128 f32
  _Float16* wh = (_Float16*)(h + (size_t)NATOM * 128);
  _Float16* Wg1t = wh;                           // 128*128
  _Float16* Wg2t = Wg1t + 128 * 128;
  _Float16* Wa1t = Wg2t + 128 * 128;             // 32*128 each below
  _Float16* Wb1t = Wa1t + 32 * 128;
  _Float16* Wn1t = Wb1t + 32 * 128;
  _Float16* Wt1t = Wn1t + 32 * 128;

  // output offsets
  float* o_kA  = out;
  float* o_eA  = out + NATOM;
  float* o_kB  = out + 2L * NATOM;
  float* o_eB  = o_kB + NBOND;
  float* o_kAn = o_eB + NBOND;
  float* o_eAn = o_kAn + NANG;
  float* o_kT  = o_eAn + NANG;
  float* o_eT  = o_kT + NTOR;
  float* o_s14 = o_eT + NTOR;
  float* o_e14 = o_s14 + NOF;
  float* o_sNB = o_e14 + NOF;
  float* o_eNB = o_sNB + NNBP;

  auto blocks = [](long n, int b) { return (unsigned)((n + b - 1) / b); };

  // 1) weight prep: f32 KxN -> f16 NxK
  transcvt<<<blocks(128L * 128, 256), 256, 0, stream>>>(gW1, Wg1t, 128);
  transcvt<<<blocks(128L * 128, 256), 256, 0, stream>>>(gW2, Wg2t, 128);
  transcvt<<<blocks(128L * 32, 256), 256, 0, stream>>>(aW1, Wa1t, 32);
  transcvt<<<blocks(128L * 32, 256), 256, 0, stream>>>(bW1, Wb1t, 32);
  transcvt<<<blocks(128L * 32, 256), 256, 0, stream>>>(nW1, Wn1t, 32);
  transcvt<<<blocks(128L * 32, 256), 256, 0, stream>>>(tW1, Wt1t, 32);

  // 2) layer 0: one-hot counts -> h = relu(cnt@gW0+gb0)
  zerof<<<blocks((long)NATOM * NTYPES, 256), 256, 0, stream>>>(m, (long)NATOM * NTYPES);
  scatter_count<<<blocks(NEDGE, 256), 256, 0, stream>>>(atom_type, edge_src, edge_dst, m);
  layer0<<<NATOM, 128, 0, stream>>>(m, gW0, gb0, h);

  // 3) layer 1: m = scatter(h); h = relu(m@gW1+gb1)
  zerof<<<blocks((long)NATOM * 128, 256), 256, 0, stream>>>(m, (long)NATOM * 128);
  scatter_add<<<blocks((long)NEDGE * 32, 256), 256, 0, stream>>>(h, edge_src, edge_dst, m);
  gemm_gcn<<<NATOM / 32, 256, 0, stream>>>(m, Wg1t, gb1, h, 1);

  // 4) layer 2: m = scatter(h); h = m@gW2+gb2
  zerof<<<blocks((long)NATOM * 128, 256), 256, 0, stream>>>(m, (long)NATOM * 128);
  scatter_add<<<blocks((long)NEDGE * 32, 256), 256, 0, stream>>>(h, edge_src, edge_dst, m);
  gemm_gcn<<<NATOM / 32, 256, 0, stream>>>(m, Wg2t, gb2, h, 0);

  // 5) readouts (WMMA 128->32, tanh, 32->2, abs)
  readout<<<blocks(NATOM, 128), 256, 0, stream>>>(h, nullptr, nullptr, 1, NATOM,
                                                  Wa1t, ab1, aW2, ab2, o_kA, o_eA);
  readout<<<blocks(NBOND, 128), 256, 0, stream>>>(h, bond_atoms, nullptr, 2, NBOND,
                                                  Wb1t, bb1, bW2, bb2, o_kB, o_eB);
  readout<<<blocks(NANG, 128), 256, 0, stream>>>(h, ang_center, ang_sides, 3, NANG,
                                                 Wn1t, nb1, nW2, nb2, o_kAn, o_eAn);
  readout<<<blocks(NTOR, 128), 256, 0, stream>>>(h, tor_atoms, nullptr, 4, NTOR,
                                                 Wt1t, tb1, tW2, tb2, o_kT, o_eT);

  // 6) pair mixing from k_atom / eq_atom already in d_out
  pair_lj<<<blocks(NOF, 256), 256, 0, stream>>>(o_kA, o_eA, of_atoms, NOF, o_s14, o_e14);
  pair_lj<<<blocks(NNBP, 256), 256, 0, stream>>>(o_kA, o_eA, nb_atoms, NNBP, o_sNB, o_eNB);
}
